// SelfAttentionLayer1D_70901320122848
// MI455X (gfx1250) — matrix operand order repaired and checked by reading, not compiled, not run on hardware
//
#include <hip/hip_runtime.h>
#include <hip/hip_bf16.h>
#include <math.h>

#define BB 8
#define NN 2048
#define CC 256
#define FF 128
#define BN (BB*NN)

typedef __bf16 bf16_t;
typedef __attribute__((ext_vector_type(16))) __bf16 v16bf;
typedef __attribute__((ext_vector_type(8)))  __bf16 v8bf;
typedef __attribute__((ext_vector_type(8)))  float  v8f;
typedef __attribute__((ext_vector_type(4)))  float  f32x4;

static __device__ __forceinline__ v16bf pack2(v8bf lo, v8bf hi) {
  v16bf r;
#pragma unroll
  for (int e = 0; e < 8; ++e) { r[e] = lo[e]; r[e + 8] = hi[e]; }
  return r;
}

// A operand (16x32 bf16, row-major source): lane row = m0 + (lane&15),
// K elements: runs [k0+hi*8, +8) and [k0+hi*8+16, +8)
static __device__ __forceinline__ v16bf load_a_bf16(const bf16_t* rowp, int k0, int hi) {
  v8bf r0 = *(const v8bf*)(rowp + k0 + hi * 8);
  v8bf r1 = *(const v8bf*)(rowp + k0 + hi * 8 + 16);
  return pack2(r0, r1);
}

// B operand (32x16 bf16): lane col = n0 + (lane&15); K runs [k0+hi*8, +8)
// and [k0+hi*8+16, +8) for transposed [N][K] storage.
static __device__ __forceinline__ v16bf load_b_bf16(const bf16_t* rowp, int k0, int hi) {
  v8bf r0 = *(const v8bf*)(rowp + k0 + hi * 8);
  v8bf r1 = *(const v8bf*)(rowp + k0 + hi * 8 + 16);
  return pack2(r0, r1);
}

static __device__ __forceinline__ v8f wmma_bf16(v16bf a, v16bf b, v8f c) {
  return __builtin_amdgcn_wmma_f32_16x16x32_bf16(false, a, false, b, (short)0, c, false, false);
}

// ---------------------------------------------------------------------------
// CDNA5 async copy global->LDS (ASYNCcnt-tracked). One instruction moves
// 32 lanes x 16B = 512B. lds32 = workgroup-relative LDS byte address (low 32
// bits of the generic pointer), g = per-lane 64-bit global address; the
// instruction's IOFFSET is added to BOTH sides (ISA 08_async_tensor §4.4).
// ---------------------------------------------------------------------------
#define ASYNC_LDS_CP(lds32, g, OFFSTR)                                        \
  asm volatile("global_load_async_to_lds_b128 %0, %1, off offset:" OFFSTR     \
               :: "v"(lds32), "v"(g) : "memory")

#if defined(__has_builtin)
#if __has_builtin(__builtin_amdgcn_s_wait_asynccnt)
#define WAIT_ASYNC(n) __builtin_amdgcn_s_wait_asynccnt(n)
#endif
#endif
#ifndef WAIT_ASYNC
#define WAIT_ASYNC(n) asm volatile("s_wait_asynccnt " #n ::: "memory")
#endif

// copy 8KB (32 rows x 256B): 16 issues of 512B
static __device__ __forceinline__ void async_copy_8k(unsigned lds32, unsigned long long g) {
  ASYNC_LDS_CP(lds32, g, "0");    ASYNC_LDS_CP(lds32, g, "512");
  ASYNC_LDS_CP(lds32, g, "1024"); ASYNC_LDS_CP(lds32, g, "1536");
  ASYNC_LDS_CP(lds32, g, "2048"); ASYNC_LDS_CP(lds32, g, "2560");
  ASYNC_LDS_CP(lds32, g, "3072"); ASYNC_LDS_CP(lds32, g, "3584");
  ASYNC_LDS_CP(lds32, g, "4096"); ASYNC_LDS_CP(lds32, g, "4608");
  ASYNC_LDS_CP(lds32, g, "5120"); ASYNC_LDS_CP(lds32, g, "5632");
  ASYNC_LDS_CP(lds32, g, "6144"); ASYNC_LDS_CP(lds32, g, "6656");
  ASYNC_LDS_CP(lds32, g, "7168"); ASYNC_LDS_CP(lds32, g, "7680");
}
// copy 4KB (16 rows x 256B): 8 issues of 512B
static __device__ __forceinline__ void async_copy_4k(unsigned lds32, unsigned long long g) {
  ASYNC_LDS_CP(lds32, g, "0");    ASYNC_LDS_CP(lds32, g, "512");
  ASYNC_LDS_CP(lds32, g, "1024"); ASYNC_LDS_CP(lds32, g, "1536");
  ASYNC_LDS_CP(lds32, g, "2048"); ASYNC_LDS_CP(lds32, g, "2560");
  ASYNC_LDS_CP(lds32, g, "3072"); ASYNC_LDS_CP(lds32, g, "3584");
}

static __device__ __forceinline__ unsigned lds_addr32(const void* p) {
  return (unsigned)(size_t)p;  // generic LDS pointer: low 32 bits = LDS offset
}

// ---------------------------------------------------------------------------
// Kernel 0: convert weights to bf16, transposed ([out_col][k] layout)
// ---------------------------------------------------------------------------
__global__ void convert_weights_kernel(const float* __restrict__ Wf, const float* __restrict__ Wg,
                                       const float* __restrict__ Wh, const float* __restrict__ Wv,
                                       bf16_t* __restrict__ Wft, bf16_t* __restrict__ Wgt,
                                       bf16_t* __restrict__ Wht, bf16_t* __restrict__ Wvt) {
  int t = blockIdx.x * blockDim.x + threadIdx.x;
  if (t < FF * CC) {
    int f = t / CC, c = t % CC;
    Wft[t] = (bf16_t)Wf[c * FF + f];
    Wgt[t] = (bf16_t)Wg[c * FF + f];
    Wht[t] = (bf16_t)Wh[c * FF + f];
  }
  if (t < FF * FF) {
    int n = t / FF, k = t % FF;
    Wvt[t] = (bf16_t)Wv[k * FF + n];
  }
}

// ---------------------------------------------------------------------------
// Kernel 1: projections fx/gx/hx = x @ W{f,g,h} + b  (bf16 out)
// one wave = 16 rows x 128 cols of one projection; K=256 in 8 steps
// ---------------------------------------------------------------------------
__global__ void proj_kernel(const float* __restrict__ x,
                            const bf16_t* __restrict__ Wft, const bf16_t* __restrict__ Wgt,
                            const bf16_t* __restrict__ Wht,
                            const float* __restrict__ bfv, const float* __restrict__ bgv,
                            const float* __restrict__ bhv,
                            bf16_t* __restrict__ fxb, bf16_t* __restrict__ gxb,
                            bf16_t* __restrict__ hxb) {
  int wid  = blockIdx.x * (blockDim.x >> 5) + (threadIdx.x >> 5);
  int lane = threadIdx.x & 31;
  int hi = lane >> 4, mrow = lane & 15;
  int p     = wid / (BN / 16);
  int strip = wid % (BN / 16);
  int row0  = strip * 16;
  const bf16_t* Wt   = (p == 0) ? Wft : (p == 1) ? Wgt : Wht;
  const float*  bias = (p == 0) ? bfv : (p == 1) ? bgv : bhv;
  bf16_t*       outb = (p == 0) ? fxb : (p == 1) ? gxb : hxb;

  v8f acc[8];
#pragma unroll
  for (int t = 0; t < 8; ++t)
#pragma unroll
    for (int v = 0; v < 8; ++v) acc[t][v] = 0.0f;

  const float* xr = x + (size_t)(row0 + mrow) * CC;
#pragma unroll
  for (int kk = 0; kk < CC / 32; ++kk) {
    int k0 = kk * 32;
    f32x4 a0 = *(const f32x4*)(xr + k0 + hi * 8);
    f32x4 a1 = *(const f32x4*)(xr + k0 + hi * 8 + 4);
    f32x4 a2 = *(const f32x4*)(xr + k0 + hi * 8 + 16);
    f32x4 a3 = *(const f32x4*)(xr + k0 + hi * 8 + 20);
    v16bf a;
#pragma unroll
    for (int e = 0; e < 4; ++e) {
      a[e] = (bf16_t)a0[e]; a[e + 4]  = (bf16_t)a1[e];
      a[e + 8] = (bf16_t)a2[e]; a[e + 12] = (bf16_t)a3[e];
    }
#pragma unroll
    for (int t = 0; t < 8; ++t) {
      const bf16_t* wr = Wt + (size_t)(t * 16 + mrow) * CC;
      v16bf bop = load_b_bf16(wr, k0, hi);
      acc[t] = wmma_bf16(a, bop, acc[t]);
    }
  }
#pragma unroll
  for (int t = 0; t < 8; ++t) {
    float bval = bias[t * 16 + mrow];
#pragma unroll
    for (int v = 0; v < 8; ++v) {
      float val = acc[t][v] + bval;
      outb[(size_t)(row0 + v + 8 * hi) * FF + t * 16 + mrow] = (bf16_t)val;
    }
  }
}

// ---------------------------------------------------------------------------
// Kernel 2: per-row softmax stats over S = fx @ gx^T, online per-lane partials
// combined with shfl_xor; writes h2t[b][f][n] = hx * exp(-M_n)/Z_n.
// gx key tiles streamed global->LDS with async copies, double-buffered.
// One wave (block of 32) per 16 query rows.
// ---------------------------------------------------------------------------
__global__ void attn_stats_kernel(const bf16_t* __restrict__ fxb, const bf16_t* __restrict__ gxb,
                                  const bf16_t* __restrict__ hxb, bf16_t* __restrict__ h2t) {
  int strip = blockIdx.x;
  int lane  = threadIdx.x & 31;
  int hi = lane >> 4, mrow = lane & 15;
  int row0 = strip * 16;
  int b    = row0 / NN;
  int nb   = row0 - b * NN;

  __shared__ __align__(16) char lds_gx[2][4096];  // double-buffered 16x128 bf16 key tile
  unsigned lbuf0 = lds_addr32(&lds_gx[0][0]) + lane * 16;
  unsigned lbuf1 = lds_addr32(&lds_gx[1][0]) + lane * 16;
  const char* gx_bytes = (const char*)(gxb + (size_t)b * NN * FF) + lane * 16;

  // prologue: stream key tile 0 into buffer 0
  async_copy_4k(lbuf0, (unsigned long long)(size_t)(gx_bytes));

  v16bf aops[4];
  const bf16_t* frow = fxb + (size_t)(row0 + mrow) * FF;
#pragma unroll
  for (int kk = 0; kk < 4; ++kk) aops[kk] = load_a_bf16(frow, kk * 32, hi);

  float mx[8], sm[8];
#pragma unroll
  for (int v = 0; v < 8; ++v) { mx[v] = -1e30f; sm[v] = 0.0f; }

  for (int mt = 0; mt < NN / 16; ++mt) {
    if (mt + 1 < NN / 16) {
      unsigned ldst = ((mt + 1) & 1) ? lbuf1 : lbuf0;
      async_copy_4k(ldst, (unsigned long long)(size_t)(gx_bytes + (size_t)(mt + 1) * 4096));
      WAIT_ASYNC(8);   // tile mt complete (8 newer copies still in flight)
    } else {
      WAIT_ASYNC(0);
    }
    const bf16_t* gtile = (const bf16_t*)&lds_gx[mt & 1][0];
    const bf16_t* grow  = gtile + mrow * FF;   // B operand from LDS (ds_load_b128)
    v8f s;
#pragma unroll
    for (int v = 0; v < 8; ++v) s[v] = 0.0f;
#pragma unroll
    for (int kk = 0; kk < 4; ++kk) {
      v16bf bop = load_b_bf16(grow, kk * 32, hi);
      s = wmma_bf16(aops[kk], bop, s);
    }
#pragma unroll
    for (int v = 0; v < 8; ++v) {
      float sv = s[v];
      float nm = fmaxf(mx[v], sv);
      sm[v] = sm[v] * __expf(mx[v] - nm) + __expf(sv - nm);
      mx[v] = nm;
    }
  }
  // combine the 16 column-lanes of each half-wave (rows v / v+8)
#pragma unroll
  for (int v = 0; v < 8; ++v) {
#pragma unroll
    for (int off = 1; off < 16; off <<= 1) {
      float om = __shfl_xor(mx[v], off, 32);
      float os = __shfl_xor(sm[v], off, 32);
      float nm = fmaxf(mx[v], om);
      sm[v] = sm[v] * __expf(mx[v] - nm) + os * __expf(om - nm);
      mx[v] = nm;
    }
  }
  __shared__ float cbuf[16];
  if (mrow == 0) {
#pragma unroll
    for (int v = 0; v < 8; ++v) cbuf[v + 8 * hi] = __expf(-mx[v]) / sm[v];
  }
  __syncthreads();
  float cn = cbuf[mrow];  // 1 / sum_m exp(S[n,m]) for row n = row0+mrow

  const bf16_t* hrow = hxb + (size_t)(row0 + mrow) * FF;
#pragma unroll
  for (int i = 0; i < 64; ++i) {
    int f = hi * 64 + i;
    float val = (float)hrow[f] * cn;
    h2t[((size_t)b * FF + f) * NN + nb + mrow] = (bf16_t)val;
  }
}

// ---------------------------------------------------------------------------
// Kernel 3: o[b][f][m] = sum_n h2[n,f] * exp(S[n,m]); one wave per (b, m-tile).
// fx query chunks (32 rows) streamed global->LDS async, double-buffered;
// S D-layout -> P B-operand re-layout via LDS round trip.
// ---------------------------------------------------------------------------
__global__ void attn_apply_kernel(const bf16_t* __restrict__ fxb, const bf16_t* __restrict__ gxb,
                                  const bf16_t* __restrict__ h2t, bf16_t* __restrict__ obb) {
  int wid  = blockIdx.x;
  int lane = threadIdx.x & 31;
  int hi = lane >> 4, mrow = lane & 15;
  int b  = wid / (NN / 16);
  int mt = wid % (NN / 16);
  int m0 = mt * 16;

  __shared__ __align__(16) char lds_fx[2][8192];  // double-buffered 32x128 bf16 query chunk
  __shared__ float sP[32 * 16];
  unsigned lbuf0 = lds_addr32(&lds_fx[0][0]) + lane * 16;
  unsigned lbuf1 = lds_addr32(&lds_fx[1][0]) + lane * 16;
  const char* fx_bytes = (const char*)(fxb + (size_t)b * NN * FF) + lane * 16;

  // prologue: stream query chunk 0 into buffer 0
  async_copy_8k(lbuf0, (unsigned long long)(size_t)(fx_bytes));

  v16bf bg[4];  // gx^T B-operands for this key tile, reused for all query chunks
  const bf16_t* grow = gxb + (size_t)(b * NN + m0 + mrow) * FF;
#pragma unroll
  for (int kk = 0; kk < 4; ++kk) bg[kk] = load_b_bf16(grow, kk * 32, hi);

  v8f oacc[8];
#pragma unroll
  for (int t = 0; t < 8; ++t)
#pragma unroll
    for (int v = 0; v < 8; ++v) oacc[t][v] = 0.0f;

  for (int nc = 0; nc < NN / 32; ++nc) {
    int n0 = nc * 32;
    if (nc + 1 < NN / 32) {
      unsigned ldst = ((nc + 1) & 1) ? lbuf1 : lbuf0;
      async_copy_8k(ldst, (unsigned long long)(size_t)(fx_bytes + (size_t)(nc + 1) * 8192));
      WAIT_ASYNC(16);  // chunk nc complete (16 newer copies still in flight)
    } else {
      WAIT_ASYNC(0);
    }
    const bf16_t* fchunk = (const bf16_t*)&lds_fx[nc & 1][0];
#pragma unroll
    for (int tile = 0; tile < 2; ++tile) {
      const bf16_t* frow = fchunk + (tile * 16 + mrow) * FF;  // A from LDS
      v8f s;
#pragma unroll
      for (int v = 0; v < 8; ++v) s[v] = 0.0f;
#pragma unroll
      for (int kk = 0; kk < 4; ++kk) {
        v16bf aop = load_a_bf16(frow, kk * 32, hi);
        s = wmma_bf16(aop, bg[kk], s);
      }
#pragma unroll
      for (int v = 0; v < 8; ++v)
        sP[(tile * 16 + v + 8 * hi) * 16 + mrow] = s[v];
    }
    __syncthreads();
    v16bf pbf;  // B-operand: k = n_local = hi*8+(e&7)+(e>>3)*16, col = mrow
#pragma unroll
    for (int e = 0; e < 16; ++e) {
      float sv = sP[(hi * 8 + (e & 7) + (e >> 3) * 16) * 16 + mrow];
      pbf[e] = (bf16_t)__expf(sv);
    }
    __syncthreads();
#pragma unroll
    for (int ft = 0; ft < 8; ++ft) {
      const bf16_t* h2row = h2t + ((size_t)b * FF + ft * 16 + mrow) * NN + n0;
      v16bf ah = load_a_bf16(h2row, 0, hi);
      oacc[ft] = wmma_bf16(ah, pbf, oacc[ft]);
    }
  }
#pragma unroll
  for (int ft = 0; ft < 8; ++ft)
#pragma unroll
    for (int v = 0; v < 8; ++v) {
      int f = ft * 16 + v + 8 * hi;
      obb[((size_t)b * FF + f) * NN + m0 + mrow] = (bf16_t)oacc[ft][v];
    }
}

// ---------------------------------------------------------------------------
// Kernel 4: y = reshape(o) @ Wv + bv. obb's [B][F][N] flat layout IS the
// reference reshape viewed as [BN][128] row-major.
// ---------------------------------------------------------------------------
__global__ void out_proj_kernel(const bf16_t* __restrict__ obb, const bf16_t* __restrict__ Wvt,
                                const float* __restrict__ bvv, float* __restrict__ out) {
  int wid  = blockIdx.x * (blockDim.x >> 5) + (threadIdx.x >> 5);
  int lane = threadIdx.x & 31;
  int hi = lane >> 4, mrow = lane & 15;
  int row0 = wid * 16;

  v8f acc[8];
#pragma unroll
  for (int t = 0; t < 8; ++t)
#pragma unroll
    for (int v = 0; v < 8; ++v) acc[t][v] = 0.0f;

  const bf16_t* rrow = obb + (size_t)(row0 + mrow) * FF;
#pragma unroll
  for (int kk = 0; kk < 4; ++kk) {
    v16bf a = load_a_bf16(rrow, kk * 32, hi);
#pragma unroll
    for (int t = 0; t < 8; ++t) {
      v16bf bop = load_b_bf16(Wvt + (size_t)(t * 16 + mrow) * FF, kk * 32, hi);
      acc[t] = wmma_bf16(a, bop, acc[t]);
    }
  }
#pragma unroll
  for (int t = 0; t < 8; ++t) {
    float bval = bvv[t * 16 + mrow];
#pragma unroll
    for (int v = 0; v < 8; ++v)
      out[(size_t)(row0 + v + 8 * hi) * FF + t * 16 + mrow] = acc[t][v] + bval;
  }
}

// ---------------------------------------------------------------------------
extern "C" void kernel_launch(void* const* d_in, const int* in_sizes, int n_in,
                              void* d_out, int out_size, void* d_ws, size_t ws_size,
                              hipStream_t stream) {
  const float* x   = (const float*)d_in[0];
  const float* Wf  = (const float*)d_in[1];
  const float* bfv = (const float*)d_in[2];
  const float* Wg  = (const float*)d_in[3];
  const float* bgv = (const float*)d_in[4];
  const float* Wh  = (const float*)d_in[5];
  const float* bhv = (const float*)d_in[6];
  const float* Wv  = (const float*)d_in[7];
  const float* bvv = (const float*)d_in[8];
  float* out = (float*)d_out;

  char* ws = (char*)d_ws;
  const size_t MB = 1024 * 1024;
  bf16_t* fxb = (bf16_t*)(ws);                 // [BN][F] bf16, 4 MiB
  bf16_t* gxb = (bf16_t*)(ws + 4 * MB);        // [BN][F]
  bf16_t* hxb = (bf16_t*)(ws + 8 * MB);        // [BN][F]
  bf16_t* h2t = (bf16_t*)(ws + 12 * MB);       // [B][F][N] scaled hx^T
  bf16_t* obb = (bf16_t*)(ws + 16 * MB);       // [B][F][N] == reshaped [BN][F]
  bf16_t* Wft = (bf16_t*)(ws + 20 * MB);       // [F][C] transposed weights
  bf16_t* Wgt = Wft + FF * CC;
  bf16_t* Wht = Wgt + FF * CC;
  bf16_t* Wvt = Wht + FF * CC;                 // [F][F]

  convert_weights_kernel<<<(FF * CC + 255) / 256, 256, 0, stream>>>(
      Wf, Wg, Wh, Wv, Wft, Wgt, Wht, Wvt);

  proj_kernel<<<(3 * (BN / 16)) / 4, 128, 0, stream>>>(
      x, Wft, Wgt, Wht, bfv, bgv, bhv, fxb, gxb, hxb);

  attn_stats_kernel<<<BN / 16, 32, 0, stream>>>(fxb, gxb, hxb, h2t);

  attn_apply_kernel<<<BB * (NN / 16), 32, 0, stream>>>(fxb, gxb, h2t, obb);

  out_proj_kernel<<<(BN / 16) / 4, 128, 0, stream>>>(obb, Wvt, bvv, out);
}
